// Net_11458972746335
// MI455X (gfx1250) — compile-verified, matrix operand order checked by reading, not run on hardware
//
#include <hip/hip_runtime.h>
#include <math.h>
#include <stdint.h>

// ---------------------------------------------------------------------------
// Types for CDNA5 WMMA (gfx1250, wave32)
// ---------------------------------------------------------------------------
typedef __attribute__((ext_vector_type(16))) __bf16 bf16x16;
typedef __attribute__((ext_vector_type(8)))  __bf16 bf16x8;
typedef __attribute__((ext_vector_type(8)))  float  f32x8;
typedef uint32_t u32x4 __attribute__((ext_vector_type(4)));
typedef uint32_t u32x8 __attribute__((ext_vector_type(8)));

union BF16Frag { bf16x16 v; bf16x8 h[2]; };

#define LEAKY 0.2f
#define BN_EPS 1e-5f
#define KNB 16

static inline int cdiv(int a, int b) { return (a + b - 1) / b; }

// ---------------------------------------------------------------------------
// f32 -> bf16 conversion with K zero-padded to a multiple of 32
// ---------------------------------------------------------------------------
__global__ void cvt_pad_bf16(const float* __restrict__ in, __bf16* __restrict__ out,
                             int M, int K, int Kp) {
  int t = blockIdx.x * blockDim.x + threadIdx.x;
  if (t >= M * Kp) return;
  int r = t / Kp, k = t % Kp;
  out[t] = (__bf16)((k < K) ? in[(size_t)r * K + k] : 0.0f);
}

__global__ void cvt_pad_bf16_t(const float* __restrict__ in, __bf16* __restrict__ out,
                               int K, int N, int Kp) {
  int t = blockIdx.x * blockDim.x + threadIdx.x;
  if (t >= N * Kp) return;
  int nIdx = t / Kp, k = t % Kp;
  out[t] = (__bf16)((k < K) ? in[(size_t)k * N + nIdx] : 0.0f);
}

// ---------------------------------------------------------------------------
// TDM: one 2-D descriptor loads a rows x 32(bf16) tile into LDS with the
// 112B row stride produced via pad_interval=16DW / pad_amount=12DW.
// 2-group D# (tile is 2-D, VADDR2/3 null). Issued per-wave; EXEC ignored.
// ---------------------------------------------------------------------------
__device__ __forceinline__ void tdm_load_tile(uint32_t ldsAddr, const __bf16* g,
                                              int rows, int strideElems) {
  uint64_t ga = (uint64_t)(uintptr_t)g;
  u32x4 g0;
  g0[0] = 1u;                                           // count=1, user D#
  g0[1] = ldsAddr;                                      // lds_addr [63:32]
  g0[2] = (uint32_t)ga;                                 // global_addr lo
  g0[3] = (uint32_t)((ga >> 32) & 0x01FFFFFFu)          // global_addr [56:32]
          | (2u << 30);                                 // type=2 ("image")
  uint32_t td0 = 32u, td1 = (uint32_t)rows;
  u32x8 g1;
  g1[0] = (1u << 16)            // data_size = 2 bytes
        | (1u << 20)            // pad_enable
        | (3u << 22)            // pad_interval: 16 DWORDs (64B row)
        | (11u << 25);          // pad_amount: 12 DWORDs (48B pad) -> 112B stride
  g1[1] = (td0 & 0xFFFFu) << 16;                        // tensor_dim0 lo16 @ [63:48]
  g1[2] = (td0 >> 16) | ((td1 & 0xFFFFu) << 16);        // td0 hi / tensor_dim1 lo
  g1[3] = (td1 >> 16) | (32u << 16);                    // td1 hi / tile_dim0=32
  g1[4] = td1 & 0xFFFFu;                                // tile_dim1=rows, tile_dim2=0
  g1[5] = (uint32_t)strideElems;                        // tensor_dim0_stride lo32
  g1[6] = 0u;                                           // stride hi16 / dim1_stride lo
  g1[7] = 0u;
  asm volatile("tensor_load_to_lds %0, %1" :: "s"(g0), "s"(g1) : "memory");
}

// ---------------------------------------------------------------------------
// bf16 WMMA GEMM:  C[M,N] = A[M,K] @ Bt[N,K]^T (+ bias[N]);  K % 32 == 0.
// 128x64 tile, BK=32, 128 threads (4 waves); wave -> 32 rows x 64 cols.
// Interior tiles: TDM tensor_load_to_lds, double-buffered, pipelined on
// TENSORcnt. Edge tiles: async-LDS b128 copies / scalar guarded staging.
// ---------------------------------------------------------------------------
#define BM 128
#define BN 64
#define BK 32
#define LDK 56   // padded K stride in LDS (bf16): 112 bytes = 7*16B

__global__ __launch_bounds__(128)
void gemm_bf16_wmma(const __bf16* __restrict__ A, const __bf16* __restrict__ Bt,
                    const float* __restrict__ bias, float* __restrict__ C,
                    int M, int N, int K) {
  __shared__ __attribute__((aligned(16))) __bf16 As[2][BM * LDK];
  __shared__ __attribute__((aligned(16))) __bf16 Bs[2][BN * LDK];

  const int tid  = threadIdx.x;
  const int wave = tid >> 5;
  const int lane = tid & 31;
  const int lm   = lane & 15;
  const int lh   = (lane >> 4) & 1;
  const int tM   = blockIdx.y * BM;
  const int tN   = blockIdx.x * BN;
  const bool fastA = (tM + BM <= M);
  const bool fastB = (tN + BN <= N);
  const bool dbl   = fastA && fastB;

  f32x8 acc0[4], acc1[4];
#pragma unroll
  for (int j = 0; j < 4; j++)
#pragma unroll
    for (int r = 0; r < 8; r++) { acc0[j][r] = 0.0f; acc1[j][r] = 0.0f; }

  auto stage_tdm = [&](int buf, int k0) {
    tdm_load_tile((uint32_t)(uintptr_t)(&As[buf][0]), A + (size_t)tM * K + k0, BM, K);
    tdm_load_tile((uint32_t)(uintptr_t)(&Bs[buf][0]), Bt + (size_t)tN * K + k0, BN, K);
  };

  auto stage_fb = [&](int buf, int k0) {
    if (fastA) {       // async: 4 x 16B per thread
#pragma unroll
      for (int it = 0; it < 4; it++) {
        int c = it * 128 + tid;
        int row = c >> 2, q = c & 3;
        const __bf16* g = A + (size_t)(tM + row) * K + k0 + q * 8;
        uint32_t l = (uint32_t)(uintptr_t)(&As[buf][row * LDK + q * 8]);
        asm volatile("global_load_async_to_lds_b128 %0, %1, off"
                     :: "v"(l), "v"(g) : "memory");
      }
    } else {
#pragma unroll
      for (int it = 0; it < 32; it++) {
        int linear = it * 128 + tid;
        int row = linear >> 5, kc = linear & 31;
        int gr = tM + row;
        __bf16 v = (__bf16)0.0f;
        if (gr < M) v = A[(size_t)gr * K + k0 + kc];
        As[buf][row * LDK + kc] = v;
      }
    }
    if (fastB) {       // async: 2 x 16B per thread
#pragma unroll
      for (int it = 0; it < 2; it++) {
        int c = it * 128 + tid;
        int row = c >> 2, q = c & 3;
        const __bf16* g = Bt + (size_t)(tN + row) * K + k0 + q * 8;
        uint32_t l = (uint32_t)(uintptr_t)(&Bs[buf][row * LDK + q * 8]);
        asm volatile("global_load_async_to_lds_b128 %0, %1, off"
                     :: "v"(l), "v"(g) : "memory");
      }
    } else {
#pragma unroll
      for (int it = 0; it < 16; it++) {
        int linear = it * 128 + tid;
        int row = linear >> 5, kc = linear & 31;
        int gr = tN + row;
        __bf16 v = (__bf16)0.0f;
        if (gr < N) v = Bt[(size_t)gr * K + k0 + kc];
        Bs[buf][row * LDK + kc] = v;
      }
    }
  };

  auto compute = [&](int buf) {
    BF16Frag af0, af1;
    const __bf16* a0 = &As[buf][(wave * 32 + lm) * LDK + 8 * lh];
    af0.h[0] = *(const bf16x8*)(a0);
    af0.h[1] = *(const bf16x8*)(a0 + 16);
    af1.h[0] = *(const bf16x8*)(a0 + 16 * LDK);
    af1.h[1] = *(const bf16x8*)(a0 + 16 * LDK + 16);
#pragma unroll
    for (int j = 0; j < 4; j++) {
      BF16Frag bf;
      const __bf16* bp = &Bs[buf][(j * 16 + lm) * LDK + 8 * lh];
      bf.h[0] = *(const bf16x8*)(bp);
      bf.h[1] = *(const bf16x8*)(bp + 16);
      acc0[j] = __builtin_amdgcn_wmma_f32_16x16x32_bf16(
          false, af0.v, false, bf.v, (short)0, acc0[j], false, false);
      acc1[j] = __builtin_amdgcn_wmma_f32_16x16x32_bf16(
          false, af1.v, false, bf.v, (short)0, acc1[j], false, false);
    }
  };

  const int nSteps = K / BK;
  if (dbl) {
    // TDM path: wave 0 issues the tile DMAs; TENSORcnt pipelining.
    if (wave == 0) stage_tdm(0, 0);
    for (int ks = 0; ks < nSteps; ks++) {
      int cur = ks & 1;
      if (wave == 0) {
        if (ks + 1 < nSteps) {
          stage_tdm(1 - cur, (ks + 1) * BK);
          __builtin_amdgcn_s_wait_tensorcnt((short)2);  // stage ks complete
        } else {
          __builtin_amdgcn_s_wait_tensorcnt((short)0);
        }
      }
      __syncthreads();
      compute(cur);
      __syncthreads();
    }
  } else {
    // Edge path: async/scalar staging, single buffer.
    for (int ks = 0; ks < nSteps; ks++) {
      stage_fb(0, ks * BK);
      asm volatile("s_wait_asynccnt 0x0" ::: "memory");
      __syncthreads();
      compute(0);
      __syncthreads();
    }
  }

  // epilogue: c[r] holds (M = r + 8*laneHalf, N = lane%16)
#pragma unroll
  for (int j = 0; j < 4; j++) {
    int col = tN + j * 16 + lm;
    if (col >= N) continue;
    float bv = bias ? bias[col] : 0.0f;
#pragma unroll
    for (int r = 0; r < 8; r++) {
      int row0 = tM + wave * 32 + r + 8 * lh;
      int row1 = row0 + 16;
      if (row0 < M) C[(size_t)row0 * N + col] = acc0[j][r] + bv;
      if (row1 < M) C[(size_t)row1 * N + col] = acc1[j][r] + bv;
    }
  }
}

// ---------------------------------------------------------------------------
// BatchNorm column stats
// ---------------------------------------------------------------------------
__global__ void col_stats(const float* __restrict__ y, float* __restrict__ mean,
                          float* __restrict__ rstd, int M, int C) {
  int c = blockIdx.x;
  __shared__ float s1[256], s2[256];
  float a = 0.f, b = 0.f;
  for (int r = threadIdx.x; r < M; r += blockDim.x) {
    float v = y[(size_t)r * C + c];
    a += v; b += v * v;
  }
  s1[threadIdx.x] = a; s2[threadIdx.x] = b;
  __syncthreads();
  for (int o = 128; o > 0; o >>= 1) {
    if ((int)threadIdx.x < o) { s1[threadIdx.x] += s1[threadIdx.x + o];
                                s2[threadIdx.x] += s2[threadIdx.x + o]; }
    __syncthreads();
  }
  if (threadIdx.x == 0) {
    float mu  = s1[0] / (float)M;
    float var = s2[0] / (float)M - mu * mu;
    mean[c] = mu;
    rstd[c] = rsqrtf(var + BN_EPS);
  }
}

__global__ void bn_act_apply(float* __restrict__ y, const float* __restrict__ mean,
                             const float* __restrict__ rstd, const float* __restrict__ g,
                             const float* __restrict__ be, float slope, int total, int C) {
  int t = blockIdx.x * blockDim.x + threadIdx.x;
  if (t >= total) return;
  int c = t % C;
  float v = g[c] * (y[t] - mean[c]) * rstd[c] + be[c];
  y[t] = (v > 0.f) ? v : v * slope;
}

// ---------------------------------------------------------------------------
// KNN (k=16, incl. self): thread-per-query with LDS-tiled candidates
// ---------------------------------------------------------------------------
__global__ __launch_bounds__(128)
void knn16(const float* __restrict__ pos, int n, int* __restrict__ idx) {
  __shared__ float spx[128], spy[128], spz[128];
  int i = blockIdx.x * 128 + threadIdx.x;
  float px = 0.f, py = 0.f, pz = 0.f;
  if (i < n) { px = pos[3 * i]; py = pos[3 * i + 1]; pz = pos[3 * i + 2]; }
  float bd[KNB]; int bi[KNB];
#pragma unroll
  for (int k = 0; k < KNB; k++) { bd[k] = 3.0e38f; bi[k] = 0; }

  for (int base = 0; base < n; base += 128) {
    __syncthreads();
    int j = base + threadIdx.x;
    if (j < n) { spx[threadIdx.x] = pos[3 * j];
                 spy[threadIdx.x] = pos[3 * j + 1];
                 spz[threadIdx.x] = pos[3 * j + 2]; }
    __syncthreads();
    int lim = (n - base < 128) ? (n - base) : 128;
    for (int t = 0; t < lim; t++) {
      float dx = px - spx[t], dy = py - spy[t], dz = pz - spz[t];
      float d = dx * dx + dy * dy + dz * dz;
      if (d < bd[KNB - 1]) {
        float dd = d; int jj = base + t;
#pragma unroll
        for (int k = 0; k < KNB; k++) {
          if (dd < bd[k]) { float td = bd[k]; int ti = bi[k];
                            bd[k] = dd; bi[k] = jj; dd = td; jj = ti; }
        }
      }
    }
  }
  if (i < n) {
#pragma unroll
    for (int k = 0; k < KNB; k++) idx[(size_t)i * KNB + k] = bi[k];
  }
}

__global__ void nn_argmin(const float* __restrict__ pf, int nf,
                          const float* __restrict__ pc, int nc, int* __restrict__ out) {
  int i = blockIdx.x * blockDim.x + threadIdx.x;
  if (i >= nf) return;
  float x = pf[3 * i], y = pf[3 * i + 1], z = pf[3 * i + 2];
  float best = 3.0e38f; int bi = 0;
  for (int j = 0; j < nc; j++) {
    float dx = x - pc[3 * j], dy = y - pc[3 * j + 1], dz = z - pc[3 * j + 2];
    float d = dx * dx + dy * dy + dz * dz;
    if (d < best) { best = d; bi = j; }
  }
  out[i] = bi;
}

// ---------------------------------------------------------------------------
// LFA helpers
// ---------------------------------------------------------------------------
__global__ void build_rel(const float* __restrict__ pos, const int* __restrict__ idx,
                          float* __restrict__ rel, int n) {
  int t = blockIdx.x * blockDim.x + threadIdx.x;
  if (t >= n * KNB) return;
  int i = t >> 4;
  int j = idx[t];
  float ix = pos[3 * i], iy = pos[3 * i + 1], iz = pos[3 * i + 2];
  float jx = pos[3 * j], jy = pos[3 * j + 1], jz = pos[3 * j + 2];
  float dx = ix - jx, dy = iy - jy, dz = iz - jz;
  float d = sqrtf(dx * dx + dy * dy + dz * dz);
  float* r = rel + (size_t)t * 10;
  r[0] = ix; r[1] = iy; r[2] = iz; r[3] = jx; r[4] = jy; r[5] = jz;
  r[6] = dx; r[7] = dy; r[8] = dz; r[9] = d;
}

__global__ void concat_feat(const float* __restrict__ h, const float* __restrict__ enc,
                            const int* __restrict__ idx, float* __restrict__ feat,
                            int n, int C) {
  int t = blockIdx.x * blockDim.x + threadIdx.x;
  int total = n * KNB * C;
  if (t >= total) return;
  int c = t % C, ik = t / C, half = C >> 1;
  float v = (c < half) ? h[(size_t)idx[ik] * half + c]
                       : enc[(size_t)ik * half + (c - half)];
  feat[t] = v;
}

__global__ void att_softmax_agg(const float* __restrict__ s, const float* __restrict__ f,
                                float* __restrict__ out, int n, int C) {
  int t = blockIdx.x * blockDim.x + threadIdx.x;
  if (t >= n * C) return;
  int i = t / C, c = t % C;
  const float* sp = s + (size_t)i * KNB * C + c;
  const float* fp = f + (size_t)i * KNB * C + c;
  float m = -3.0e38f;
#pragma unroll
  for (int k = 0; k < KNB; k++) m = fmaxf(m, sp[(size_t)k * C]);
  float e[KNB]; float sum = 0.f;
#pragma unroll
  for (int k = 0; k < KNB; k++) { e[k] = __expf(sp[(size_t)k * C] - m); sum += e[k]; }
  float inv = 1.f / sum, acc = 0.f;
#pragma unroll
  for (int k = 0; k < KNB; k++) acc += e[k] * inv * fp[(size_t)k * C];
  out[t] = acc;
}

__global__ void add_leaky(const float* __restrict__ a, const float* __restrict__ b,
                          float* __restrict__ out, int total) {
  int t = blockIdx.x * blockDim.x + threadIdx.x;
  if (t >= total) return;
  float v = a[t] + b[t];
  out[t] = (v > 0.f) ? v : v * LEAKY;
}

__global__ void gather_concat(const float* __restrict__ xc, const int* __restrict__ nn,
                              const float* __restrict__ xs, float* __restrict__ out,
                              int n, int Cc, int Cs) {
  int C = Cc + Cs;
  int t = blockIdx.x * blockDim.x + threadIdx.x;
  if (t >= n * C) return;
  int i = t / C, c = t % C;
  out[t] = (c < Cc) ? xc[(size_t)nn[i] * Cc + c] : xs[(size_t)i * Cs + (c - Cc)];
}

__global__ void log_softmax_rows(const float* __restrict__ x, float* __restrict__ out,
                                 int n, int C) {
  int i = blockIdx.x * blockDim.x + threadIdx.x;
  if (i >= n) return;
  const float* r = x + (size_t)i * C;
  float m = -3.0e38f;
  for (int c = 0; c < C; c++) m = fmaxf(m, r[c]);
  float s = 0.f;
  for (int c = 0; c < C; c++) s += __expf(r[c] - m);
  float ls = __logf(s);
  for (int c = 0; c < C; c++) out[(size_t)i * C + c] = r[c] - m - ls;
}

// ---------------------------------------------------------------------------
// Host-side orchestration
// ---------------------------------------------------------------------------
struct BnLin { const float *W, *b, *g, *be; };
struct LfaP  { BnLin enc; const float* attW; BnLin post; };
struct BlockP{ BnLin sc, m1; LfaP l1, l2; BnLin m2; };
struct HeadP { const float *W1, *b1, *g1, *be1, *W2, *b2; };
struct LinP  { const float *W, *b; };

struct Ws {
  char* base; size_t off;
  void* get(size_t bytes) {
    off = (off + 255) & ~(size_t)255;
    void* p = base + off; off += bytes; return p;
  }
  float* f(size_t n) { return (float*)get(n * sizeof(float)); }
  int*   i(size_t n) { return (int*)get(n * sizeof(int)); }
  __bf16* h(size_t n) { return (__bf16*)get(n * sizeof(__bf16)); }
};

static void gemm(const float* A, const float* Bw, const float* bias, float* C,
                 int M, int N, int K, Ws& ws, hipStream_t st) {
  int Kp = (K + 31) & ~31;
  size_t save = ws.off;
  __bf16* Ab = ws.h((size_t)M * Kp);
  __bf16* Bb = ws.h((size_t)N * Kp);
  cvt_pad_bf16<<<cdiv(M * Kp, 256), 256, 0, st>>>(A, Ab, M, K, Kp);
  cvt_pad_bf16_t<<<cdiv(N * Kp, 256), 256, 0, st>>>(Bw, Bb, K, N, Kp);
  dim3 g(cdiv(N, BN), cdiv(M, BM));
  gemm_bf16_wmma<<<g, 128, 0, st>>>(Ab, Bb, bias, C, M, N, Kp);
  ws.off = save;
}

static void smlp(const float* x, const BnLin& p, float* y, int M, int N, int K,
                 float slope, float* mean, float* rstd, Ws& ws, hipStream_t st) {
  gemm(x, p.W, p.b, y, M, N, K, ws, st);
  col_stats<<<N, 256, 0, st>>>(y, mean, rstd, M, N);
  int tot = M * N;
  bn_act_apply<<<cdiv(tot, 256), 256, 0, st>>>(y, mean, rstd, p.g, p.be, slope, tot, N);
}

static void run_lfa(const float* x, const float* pos, const int* idx,
                    const LfaP& p, int n, int ch, float* out,
                    float* mean, float* rstd, Ws& ws, hipStream_t st) {
  size_t save = ws.off;
  float* rel  = ws.f((size_t)n * KNB * 10);
  float* enc  = ws.f((size_t)n * KNB * (ch / 2));
  float* feat = ws.f((size_t)n * KNB * ch);
  float* sco  = ws.f((size_t)n * KNB * ch);
  float* agg  = ws.f((size_t)n * ch);

  build_rel<<<cdiv(n * KNB, 256), 256, 0, st>>>(pos, idx, rel, n);
  smlp(rel, p.enc, enc, n * KNB, ch / 2, 10, LEAKY, mean, rstd, ws, st);
  concat_feat<<<cdiv(n * KNB * ch, 256), 256, 0, st>>>(x, enc, idx, feat, n, ch);
  gemm(feat, p.attW, nullptr, sco, n * KNB, ch, ch, ws, st);
  att_softmax_agg<<<cdiv(n * ch, 256), 256, 0, st>>>(sco, feat, agg, n, ch);
  smlp(agg, p.post, out, n, ch, ch, LEAKY, mean, rstd, ws, st);
  ws.off = save;
}

static void run_block(const float* x, const float* pos, int n, int din, int dout,
                      const BlockP& B, float* hOut,
                      float* mean, float* rstd, Ws& ws, hipStream_t st) {
  size_t save = ws.off;
  int*   idx = ws.i((size_t)n * KNB);
  float* sc  = ws.f((size_t)n * dout);
  float* h8  = ws.f((size_t)n * (dout / 8));
  float* q1  = ws.f((size_t)n * (dout / 4));
  float* q2  = ws.f((size_t)n * (dout / 2));
  float* h2  = ws.f((size_t)n * dout);

  knn16<<<cdiv(n, 128), 128, 0, st>>>(pos, n, idx);
  smlp(x, B.sc, sc, n, dout, din, 1.0f, mean, rstd, ws, st);
  smlp(x, B.m1, h8, n, dout / 8, din, LEAKY, mean, rstd, ws, st);
  run_lfa(h8, pos, idx, B.l1, n, dout / 4, q1, mean, rstd, ws, st);
  run_lfa(q1, pos, idx, B.l2, n, dout / 2, q2, mean, rstd, ws, st);
  smlp(q2, B.m2, h2, n, dout, dout / 2, 1.0f, mean, rstd, ws, st);
  add_leaky<<<cdiv(n * dout, 256), 256, 0, st>>>(h2, sc, hOut, n * dout);
  ws.off = save;
}

static void run_head(const float* x, const HeadP& p, float* tmp, float* out,
                     int n, int d0, int d1, int d2, float* mean, float* rstd,
                     Ws& ws, hipStream_t st) {
  gemm(x, p.W1, p.b1, tmp, n, d1, d0, ws, st);
  col_stats<<<d1, 256, 0, st>>>(tmp, mean, rstd, n, d1);
  int tot = n * d1;
  bn_act_apply<<<cdiv(tot, 256), 256, 0, st>>>(tmp, mean, rstd, p.g1, p.be1,
                                               0.0f, tot, d1);
  gemm(tmp, p.W2, p.b2, out, n, d2, d1, ws, st);
}

extern "C" void kernel_launch(void* const* d_in, const int* in_sizes, int n_in,
                              void* d_out, int out_size, void* d_ws, size_t ws_size,
                              hipStream_t stream) {
  (void)in_sizes; (void)n_in; (void)out_size; (void)ws_size;
  const float* x0  = (const float*)d_in[0];   // [16384, 6]
  const float* pos = (const float*)d_in[1];   // [16384, 3]
  // d_in[2] = batch (unused)

  int pi = 3;
  auto nxt = [&]() { return (const float*)d_in[pi++]; };
  auto rdBn  = [&]() { BnLin p; p.W = nxt(); p.b = nxt(); p.g = nxt(); p.be = nxt(); return p; };
  auto rdLfa = [&]() { LfaP l; l.enc = rdBn(); l.attW = nxt(); l.post = rdBn(); return l; };

  BlockP B[4];
  for (int i = 0; i < 4; i++) {
    B[i].sc = rdBn(); B[i].m1 = rdBn();
    B[i].l1 = rdLfa(); B[i].l2 = rdLfa();
    B[i].m2 = rdBn();
  }
  HeadP hd1; hd1.W1 = nxt(); hd1.b1 = nxt(); hd1.g1 = nxt(); hd1.be1 = nxt();
             hd1.W2 = nxt(); hd1.b2 = nxt();
  LinP fp4p{nxt(), nxt()}, fp3p{nxt(), nxt()}, fp2p{nxt(), nxt()}, fp1p{nxt(), nxt()};
  HeadP hd2; hd2.W1 = nxt(); hd2.b1 = nxt(); hd2.g1 = nxt(); hd2.be1 = nxt();
             hd2.W2 = nxt(); hd2.b2 = nxt();
  LinP lin{nxt(), nxt()};

  Ws ws{(char*)d_ws, 0};
  float* mean = ws.f(512);
  float* rstd = ws.f(512);

  float* h1  = ws.f((size_t)16384 * 32);
  float* h2b = ws.f((size_t)4096 * 128);
  float* h3  = ws.f((size_t)1024 * 256);
  float* h4  = ws.f((size_t)256 * 512);
  float* m   = ws.f((size_t)64 * 512);
  float* t1  = ws.f((size_t)64 * 512);
  int*   nn4 = ws.i(256);   float* cat4 = ws.f((size_t)256 * 768);   float* f4 = ws.f((size_t)256 * 256);
  int*   nn3 = ws.i(1024);  float* cat3 = ws.f((size_t)1024 * 384);  float* f3 = ws.f((size_t)1024 * 128);
  int*   nn2 = ws.i(4096);  float* cat2 = ws.f((size_t)4096 * 160);  float* f2 = ws.f((size_t)4096 * 32);
  int*   nn1 = ws.i(16384); float* cat1 = ws.f((size_t)16384 * 38);  float* f1 = ws.f((size_t)16384 * 8);
  float* t2  = ws.f((size_t)16384 * 64);
  float* hf  = ws.f((size_t)16384 * 32);
  float* lg  = ws.f((size_t)16384 * 13);

  // ---- encoder ----
  run_block(x0,  pos, 16384,   6,  32, B[0], h1,  mean, rstd, ws, stream);
  run_block(h1,  pos,  4096,  32, 128, B[1], h2b, mean, rstd, ws, stream);
  run_block(h2b, pos,  1024, 128, 256, B[2], h3,  mean, rstd, ws, stream);
  run_block(h3,  pos,   256, 256, 512, B[3], h4,  mean, rstd, ws, stream);

  // ---- bottleneck head ----
  run_head(h4, hd1, t1, m, 64, 512, 512, 512, mean, rstd, ws, stream);

  // ---- decoder ----
  nn_argmin<<<cdiv(256, 128), 128, 0, stream>>>(pos, 256, pos, 64, nn4);
  gather_concat<<<cdiv(256 * 768, 256), 256, 0, stream>>>(m, nn4, h3, cat4, 256, 512, 256);
  gemm(cat4, fp4p.W, fp4p.b, f4, 256, 256, 768, ws, stream);

  nn_argmin<<<cdiv(1024, 128), 128, 0, stream>>>(pos, 1024, pos, 256, nn3);
  gather_concat<<<cdiv(1024 * 384, 256), 256, 0, stream>>>(f4, nn3, h2b, cat3, 1024, 256, 128);
  gemm(cat3, fp3p.W, fp3p.b, f3, 1024, 128, 384, ws, stream);

  nn_argmin<<<cdiv(4096, 128), 128, 0, stream>>>(pos, 4096, pos, 1024, nn2);
  gather_concat<<<cdiv(4096 * 160, 256), 256, 0, stream>>>(f3, nn2, h1, cat2, 4096, 128, 32);
  gemm(cat2, fp2p.W, fp2p.b, f2, 4096, 32, 160, ws, stream);

  nn_argmin<<<cdiv(16384, 128), 128, 0, stream>>>(pos, 16384, pos, 4096, nn1);
  gather_concat<<<cdiv(16384 * 38, 256), 256, 0, stream>>>(f2, nn1, x0, cat1, 16384, 32, 6);
  gemm(cat1, fp1p.W, fp1p.b, f1, 16384, 8, 38, ws, stream);

  // ---- classification head ----
  run_head(f1, hd2, t2, hf, 16384, 8, 64, 32, mean, rstd, ws, stream);
  gemm(hf, lin.W, lin.b, lg, 16384, 13, 32, ws, stream);
  log_softmax_rows<<<cdiv(16384, 256), 256, 0, stream>>>(lg, (float*)d_out, 16384, 13);
}
